// HeteroGNNWithPRMP_19808389169917
// MI455X (gfx1250) — compile-verified, matrix operand order checked by reading, not run on hardware
//
#include <hip/hip_runtime.h>
#include <cstddef>

#define CDIM 256
#define KT   32

typedef __attribute__((ext_vector_type(16))) __bf16 v16bf;
typedef __attribute__((ext_vector_type(8)))  float  v8f;

union FragAB { v16bf v; uint4 q[2]; };

__device__ __forceinline__ unsigned short f2bf(float f) {
  unsigned u = __float_as_uint(f);
  unsigned r = u + 0x7FFFu + ((u >> 16) & 1u);   // round-to-nearest-even
  return (unsigned short)(r >> 16);
}

// ---------------------------------------------------------------------------
// Fused GEMM: out = OP( A @ W + bias ), N = 256 fixed, K = 256 or 512.
// A cols 0..255 from A1; cols 256..511 from A2 * 1/max(cnt,1) (fused concat).
// OP: 0 = relu, 1 = identity, 2 = LayerNorm(g,b) + relu   (all f32 out)
// Block: 256 threads (8 waves), tile 64 rows x 256 cols, K stepped by 32,
// double-buffered LDS staging (bf16), v_wmma_f32_16x16x32_bf16 compute.
// ---------------------------------------------------------------------------
template<int OP>
__global__ __launch_bounds__(256)
void gemm_fused_kernel(const float* __restrict__ A1, const float* __restrict__ A2,
                       const float* __restrict__ cnt,
                       const unsigned short* __restrict__ W,   // bf16 [K,256] row-major
                       const float* __restrict__ bias,
                       const float* __restrict__ lng, const float* __restrict__ lnb,
                       float* __restrict__ out, int M, int K)
{
  __shared__ unsigned short lsA[2][64 * KT];       // 2 x 4 KB
  __shared__ unsigned short lsBT[2][CDIM * KT];    // 2 x 16 KB (B transposed)
  __shared__ float shs[64][2][2];                  // 1 KB per-row LN partials

  const int tid   = threadIdx.x;
  const int wave  = tid >> 5;
  const int lane  = tid & 31;
  const int half  = lane >> 4;          // K-half of the fragment
  const int lanen = lane & 15;
  const int rtile = (wave >> 1) << 4;   // 0,16,32,48
  const int chalf = wave & 1;           // column half
  const int c0    = chalf * 128;
  const int row0  = blockIdx.x * 64;

  // ---- loop-invariant staging state (one row per thread) ----
  const int  sr   = tid >> 2;                 // 0..63
  const int  scb  = (tid & 3) << 3;           // 0,8,16,24
  const int  sgr  = row0 + sr;
  const bool srOK = (sgr < M);
  float scale = 1.f;
  if (cnt != nullptr && srOK) scale = __builtin_amdgcn_rcpf(fmaxf(cnt[sgr], 1.f));
  const float* a1p = A1 + (size_t)sgr * CDIM + scb;
  const float* a2p = (A2 != nullptr) ? (A2 + (size_t)sgr * CDIM + scb) : nullptr;

  v8f acc[8];
#pragma unroll
  for (int i = 0; i < 8; ++i) acc[i] = (v8f){0.f,0.f,0.f,0.f,0.f,0.f,0.f,0.f};

  auto stageA = [&](int buf, int k0) {
    union { unsigned short s[8]; uint4 q; } t;
    if (srOK) {
      const float* sp; float sc;
      if (k0 < CDIM) { sp = a1p + k0;          sc = 1.f;   }
      else           { sp = a2p + (k0 - CDIM); sc = scale; }
      float4 f0 = *(const float4*)sp;
      float4 f1 = *(const float4*)(sp + 4);
      t.s[0] = f2bf(f0.x * sc); t.s[1] = f2bf(f0.y * sc);
      t.s[2] = f2bf(f0.z * sc); t.s[3] = f2bf(f0.w * sc);
      t.s[4] = f2bf(f1.x * sc); t.s[5] = f2bf(f1.y * sc);
      t.s[6] = f2bf(f1.z * sc); t.s[7] = f2bf(f1.w * sc);
    } else {
      t.q = (uint4){0u,0u,0u,0u};
    }
    *(uint4*)&lsA[buf][sr * KT + scb] = t.q;
  };

  auto stageB = [&](int buf, int k0) {
#pragma unroll
    for (int i = 0; i < 4; ++i) {
      int pidx = tid + i * 256;          // 0..1023
      int kk   = pidx >> 5;              // 0..31
      int n8   = (pidx & 31) << 3;       // 0..248
      union { uint4 q; unsigned short s[8]; } t;
      t.q = *(const uint4*)&W[(size_t)(k0 + kk) * CDIM + n8];
#pragma unroll
      for (int j = 0; j < 8; ++j) lsBT[buf][(n8 + j) * KT + kk] = t.s[j];
    }
  };

  auto compute = [&](int buf) {
    FragAB fa;
    const uint4* arow = (const uint4*)&lsA[buf][(rtile + lanen) * KT];
    fa.q[0] = arow[half];
    fa.q[1] = arow[half + 2];
#pragma unroll
    for (int nt = 0; nt < 8; ++nt) {
      FragAB fb;
      const uint4* bcol = (const uint4*)&lsBT[buf][(c0 + nt * 16 + lanen) * KT];
      fb.q[0] = bcol[half];
      fb.q[1] = bcol[half + 2];
      acc[nt] = __builtin_amdgcn_wmma_f32_16x16x32_bf16(
          false, fa.v, false, fb.v, (short)0, acc[nt], false, false);
    }
  };

  // ---- double-buffered K loop: one barrier per step ----
  stageA(0, 0);
  stageB(0, 0);
  __syncthreads();
  int buf = 0;
  for (int k0 = 0; k0 < K; k0 += KT) {
    int kn = k0 + KT;
    if (kn < K) { stageA(buf ^ 1, kn); stageB(buf ^ 1, kn); }
    compute(buf);
    __syncthreads();
    buf ^= 1;
  }

  // ---- epilogue ----
  if (OP <= 1) {
#pragma unroll
    for (int nt = 0; nt < 8; ++nt) {
      int col  = c0 + nt * 16 + lanen;
      float bv = bias[col];
#pragma unroll
      for (int j = 0; j < 8; ++j) {
        int gr  = row0 + rtile + (half << 3) + j;
        float v = acc[nt][j] + bv;
        if (OP == 0) v = fmaxf(v, 0.f);
        if (gr < M) out[(size_t)gr * CDIM + col] = v;
      }
    }
  } else {
    // LayerNorm over 256 cols (row split across two waves) + relu
    float sum[8], sq[8];
#pragma unroll
    for (int j = 0; j < 8; ++j) { sum[j] = 0.f; sq[j] = 0.f; }
#pragma unroll
    for (int nt = 0; nt < 8; ++nt) {
      float bv = bias[c0 + nt * 16 + lanen];
#pragma unroll
      for (int j = 0; j < 8; ++j) {
        float v = acc[nt][j] + bv;
        acc[nt][j] = v;
        sum[j] += v; sq[j] += v * v;
      }
    }
#pragma unroll
    for (int m = 1; m <= 8; m <<= 1) {
#pragma unroll
      for (int j = 0; j < 8; ++j) {
        sum[j] += __shfl_xor(sum[j], m);
        sq[j]  += __shfl_xor(sq[j],  m);
      }
    }
    if (lanen == 0) {
#pragma unroll
      for (int j = 0; j < 8; ++j) {
        int lr = rtile + (half << 3) + j;
        shs[lr][chalf][0] = sum[j];
        shs[lr][chalf][1] = sq[j];
      }
    }
    __syncthreads();
    float mu[8], inv[8];
#pragma unroll
    for (int j = 0; j < 8; ++j) {
      int lr = rtile + (half << 3) + j;
      float ts = shs[lr][0][0] + shs[lr][1][0];
      float tq = shs[lr][0][1] + shs[lr][1][1];
      float m_ = ts * (1.f / 256.f);
      float var = tq * (1.f / 256.f) - m_ * m_;
      mu[j]  = m_;
      inv[j] = rsqrtf(var + 1e-5f);
    }
#pragma unroll
    for (int nt = 0; nt < 8; ++nt) {
      int col = c0 + nt * 16 + lanen;
      float gg = lng[col], bb = lnb[col];
#pragma unroll
      for (int j = 0; j < 8; ++j) {
        int gr = row0 + rtile + (half << 3) + j;
        float v = (acc[nt][j] - mu[j]) * inv[j] * gg + bb;
        v = fmaxf(v, 0.f);
        if (gr < M) out[(size_t)gr * CDIM + col] = v;
      }
    }
  }
}

// ---------------------------------------------------------------------------
// Edge pass: residual = x_src[src] - pred[dst]; msg = LN(residual)*g+b;
// atomic scatter-add into agg[dst], count into cnt[dst].
// One wave (32 lanes) per edge, 8 channels/lane, float4 loads.
// ---------------------------------------------------------------------------
__global__ __launch_bounds__(256)
void edge_kernel(const float* __restrict__ xsrc, const float* __restrict__ pred,
                 const int* __restrict__ src, const int* __restrict__ dst,
                 const float* __restrict__ mg, const float* __restrict__ mb,
                 float* __restrict__ agg, float* __restrict__ cnt, int E)
{
  int wave = threadIdx.x >> 5;
  int lane = threadIdx.x & 31;
  int e = blockIdx.x * 8 + wave;
  if (e >= E) return;
  int s = src[e], d = dst[e];
  int cb = lane * 8;
  const float4* ps = (const float4*)&xsrc[(size_t)s * CDIM + cb];
  const float4* pp = (const float4*)&pred[(size_t)d * CDIM + cb];
  float4 a0 = ps[0], a1 = ps[1];
  float4 b0 = pp[0], b1 = pp[1];
  float r[8] = {a0.x-b0.x, a0.y-b0.y, a0.z-b0.z, a0.w-b0.w,
                a1.x-b1.x, a1.y-b1.y, a1.z-b1.z, a1.w-b1.w};
  float sum = 0.f, sq = 0.f;
#pragma unroll
  for (int j = 0; j < 8; ++j) { sum += r[j]; sq += r[j] * r[j]; }
#pragma unroll
  for (int m = 16; m >= 1; m >>= 1) { sum += __shfl_xor(sum, m); sq += __shfl_xor(sq, m); }
  float mu  = sum * (1.f / 256.f);
  float inv = rsqrtf(sq * (1.f / 256.f) - mu * mu + 1e-5f);
  float4 g0 = *(const float4*)&mg[cb];
  float4 g1 = *(const float4*)&mg[cb + 4];
  float4 h0 = *(const float4*)&mb[cb];
  float4 h1 = *(const float4*)&mb[cb + 4];
  float gv[8] = {g0.x,g0.y,g0.z,g0.w, g1.x,g1.y,g1.z,g1.w};
  float bv[8] = {h0.x,h0.y,h0.z,h0.w, h1.x,h1.y,h1.z,h1.w};
#pragma unroll
  for (int j = 0; j < 8; ++j) {
    float msg = (r[j] - mu) * inv * gv[j] + bv[j];
    atomicAdd(&agg[(size_t)d * CDIM + cb + j], msg);
  }
  if (lane == 0) atomicAdd(&cnt[d], 1.0f);
}

__global__ void zero4_kernel(float4* __restrict__ p, size_t n4) {
  size_t i = (size_t)blockIdx.x * blockDim.x + threadIdx.x;
  size_t st = (size_t)gridDim.x * blockDim.x;
  for (; i < n4; i += st) p[i] = (float4){0.f, 0.f, 0.f, 0.f};
}

__global__ void zero_kernel(float* __restrict__ p, size_t n) {
  size_t i = (size_t)blockIdx.x * blockDim.x + threadIdx.x;
  size_t st = (size_t)gridDim.x * blockDim.x;
  for (; i < n; i += st) p[i] = 0.f;
}

__global__ void cvt_bf16_kernel(const float* __restrict__ in,
                                unsigned short* __restrict__ o, size_t n) {
  size_t i = (size_t)blockIdx.x * blockDim.x + threadIdx.x;
  size_t st = (size_t)gridDim.x * blockDim.x;
  for (; i < n; i += st) o[i] = f2bf(in[i]);
}

// ---------------------------------------------------------------------------
extern "C" void kernel_launch(void* const* d_in, const int* in_sizes, int n_in,
                              void* d_out, int out_size, void* d_ws, size_t ws_size,
                              hipStream_t stream) {
  const float* x_user = (const float*)d_in[0];
  const float* x_item = (const float*)d_in[1];
  const float* pw1    = (const float*)d_in[2];
  const float* pb1    = (const float*)d_in[3];
  const float* pw2    = (const float*)d_in[4];
  const float* pb2    = (const float*)d_in[5];
  const float* msg_g  = (const float*)d_in[6];
  const float* msg_b  = (const float*)d_in[7];
  const float* lin_w  = (const float*)d_in[8];
  const float* lin_b  = (const float*)d_in[9];
  const float* node_g = (const float*)d_in[10];
  const float* node_b = (const float*)d_in[11];
  const int*   ei_ui  = (const int*)d_in[12];
  const int*   ei_iu  = (const int*)d_in[13];

  const int C  = CDIM;
  const int NU = in_sizes[0] / C;
  const int NI = in_sizes[1] / C;
  const int L  = in_sizes[3] / (2 * C);
  const int E  = in_sizes[12] / 2;
  const int Nmax = (NU > NI) ? NU : NI;

  // ---- workspace carve ----
  char* p = (char*)d_ws;
  auto carve = [&](size_t bytes) -> void* {
    void* r = (void*)p;
    p += (bytes + 255) & ~(size_t)255;
    return r;
  };
  unsigned short* pw1b = (unsigned short*)carve((size_t)L * 2 * C * C * 2);
  unsigned short* pw2b = (unsigned short*)carve((size_t)L * 2 * C * C * 2);
  unsigned short* linb = (unsigned short*)carve((size_t)L * 2 * 2 * C * C * 2);
  float* H    = (float*)carve((size_t)Nmax * C * 4);   // hidden; reused as agg
  float* pred = (float*)carve((size_t)Nmax * C * 4);
  float* cntb = (float*)carve((size_t)Nmax * 4);
  float* X1u  = (float*)carve((size_t)NU * C * 4);
  float* X1i  = (float*)carve((size_t)NI * C * 4);
  float* agg  = H;

  // ---- weight conversion fp32 -> bf16 ----
  cvt_bf16_kernel<<<512, 256, 0, stream>>>(pw1,   pw1b, (size_t)L * 2 * C * C);
  cvt_bf16_kernel<<<512, 256, 0, stream>>>(pw2,   pw2b, (size_t)L * 2 * C * C);
  cvt_bf16_kernel<<<512, 256, 0, stream>>>(lin_w, linb, (size_t)L * 2 * 2 * C * C);

  float* outF = (float*)d_out;

  // one conv: dst-MLP (per NODE, not per edge) -> edge LN scatter -> SAGE update
  auto conv = [&](const float* xsrc, const float* xdst, int Ndst,
                  const int* ei, int l, int t, int nodeT, float* outX) {
    size_t wi = (size_t)l * 2 + t;
    const unsigned short* w1 = pw1b + wi * C * C;
    const unsigned short* w2 = pw2b + wi * C * C;
    const unsigned short* wl = linb + wi * 2 * C * C;
    const float* b1 = pb1 + wi * C;
    const float* b2 = pb2 + wi * C;
    const float* lb = lin_b + wi * C;
    const float* mg = msg_g + wi * C;
    const float* mb = msg_b + wi * C;
    const float* ng = node_g + ((size_t)l * 2 + nodeT) * C;
    const float* nb = node_b + ((size_t)l * 2 + nodeT) * C;
    int gb = (Ndst + 63) / 64;
    // H = relu(x_dst @ W1 + b1)
    gemm_fused_kernel<0><<<gb, 256, 0, stream>>>(xdst, nullptr, nullptr, w1, b1,
                                                 nullptr, nullptr, H, Ndst, C);
    // pred = H @ W2 + b2
    gemm_fused_kernel<1><<<gb, 256, 0, stream>>>(H, nullptr, nullptr, w2, b2,
                                                 nullptr, nullptr, pred, Ndst, C);
    // scatter mean of LN residuals
    zero4_kernel<<<2048, 256, 0, stream>>>((float4*)agg, (size_t)Ndst * C / 4);
    zero_kernel<<<256, 256, 0, stream>>>(cntb, (size_t)Ndst);
    edge_kernel<<<(E + 7) / 8, 256, 0, stream>>>(xsrc, pred, ei, ei + E,
                                                 mg, mb, agg, cntb, E);
    // out = relu(LN(concat(x_dst, agg/cnt) @ lin_w + lin_b))
    gemm_fused_kernel<2><<<gb, 256, 0, stream>>>(xdst, agg, cntb, wl, lb,
                                                 ng, nb, outX, Ndst, 2 * C);
  };

  const float* cu = x_user;
  const float* ci = x_item;
  for (int l = 0; l < L; ++l) {
    bool toOut = (((L - 1 - l) & 1) == 0);     // final layer lands in d_out
    float* nu_ = toOut ? outF : X1u;
    float* ni_ = toOut ? (outF + (size_t)NU * C) : X1i;
    // edge type 0: user -> item (dst = item, node-LN type 1)
    conv(cu, ci, NI, ei_ui, l, 0, 1, ni_);
    // edge type 1: item -> user (dst = user, node-LN type 0)
    conv(ci, cu, NU, ei_iu, l, 1, 0, nu_);
    cu = nu_; ci = ni_;
  }
}
